// linear_channel_attention_6004364279994
// MI455X (gfx1250) — matrix operand order repaired and checked by reading, not compiled, not run on hardware
//
#include <hip/hip_runtime.h>

// ---------------------------------------------------------------------------
// CDNA5 / gfx1250: linear channel attention pipeline in bf16 WMMA.
//   1. LN(x) -> h (bf16)
//   2. qkv = h @ Wqkv  (WMMA bf16, fused q/k/v extraction; k,v transposed)
//   3. softmax rows (q over E, kT over N)
//   4. kv   = phiKT @ vT^T   (WMMA, batched)
//   5. attn = phiQ @ kvT     (WMMA, batched)
//   6. LN(attn) -> a (bf16); out = a @ WprojT (WMMA)
//
// GEMM: block tile 128x128, 8 waves (4Mx2N), wave tile 32x64 (8 WMMA/K-step),
// K-step 32, LDS double-buffered with register prefetch (1 barrier/K-step).
// ---------------------------------------------------------------------------

typedef __bf16 bf16;
typedef __attribute__((ext_vector_type(8)))  bf16  v8bf;
typedef __attribute__((ext_vector_type(16))) bf16  v16bf;
typedef __attribute__((ext_vector_type(8)))  float v8f;

#define LN_EPS 1e-5f
#define PAD_LD 40           // 32 K-elements + 8 pad (80B row stride, 20 banks)
#define TILE_ELEMS (128 * PAD_LD)

// ---------------- wave32 reductions -----------------
__device__ __forceinline__ float wave_sum(float v) {
#pragma unroll
  for (int o = 16; o > 0; o >>= 1) v += __shfl_xor(v, o, 32);
  return v;
}
__device__ __forceinline__ float wave_max(float v) {
#pragma unroll
  for (int o = 16; o > 0; o >>= 1) v = fmaxf(v, __shfl_xor(v, o, 32));
  return v;
}

__device__ __forceinline__ v16bf cat8(v8bf lo, v8bf hi) {
  return __builtin_shufflevector(lo, hi, 0, 1, 2, 3, 4, 5, 6, 7, 8, 9, 10, 11,
                                 12, 13, 14, 15);
}

// ---------------- LayerNorm row kernel (f32 in -> bf16 out) -----------------
__global__ __launch_bounds__(256) void ln_bf16_kernel(
    const float* __restrict__ X, const float* __restrict__ g,
    const float* __restrict__ bta, bf16* __restrict__ Y, int L) {
  const int row = blockIdx.x;
  const float* x = X + (size_t)row * L;
  bf16* y = Y + (size_t)row * L;
  const int tid = threadIdx.x, lane = tid & 31, wave = tid >> 5;
  __shared__ float s1[8], s2[8];
  float sum = 0.f, sq = 0.f;
  for (int i = tid; i < L; i += 256) { float v = x[i]; sum += v; sq += v * v; }
  sum = wave_sum(sum); sq = wave_sum(sq);
  if (lane == 0) { s1[wave] = sum; s2[wave] = sq; }
  __syncthreads();
  float tot = 0.f, totq = 0.f;
#pragma unroll
  for (int w = 0; w < 8; ++w) { tot += s1[w]; totq += s2[w]; }
  const float mean = tot / (float)L;
  const float var  = totq / (float)L - mean * mean;
  const float rstd = rsqrtf(var + LN_EPS);
  for (int i = tid; i < L; i += 256)
    y[i] = (bf16)((x[i] - mean) * rstd * g[i] + bta[i]);
}

// ---------------- Softmax row kernel (f32 in -> bf16 out) -----------------
__global__ __launch_bounds__(256) void softmax_bf16_kernel(
    const float* __restrict__ X, bf16* __restrict__ Y, int L) {
  const int row = blockIdx.x;
  const float* x = X + (size_t)row * L;
  bf16* y = Y + (size_t)row * L;
  const int tid = threadIdx.x, lane = tid & 31, wave = tid >> 5;
  __shared__ float sm[8], ss[8];
  float mx = -3.4e38f;
  for (int i = tid; i < L; i += 256) mx = fmaxf(mx, x[i]);
  mx = wave_max(mx);
  if (lane == 0) sm[wave] = mx;
  __syncthreads();
  float m = sm[0];
#pragma unroll
  for (int w = 1; w < 8; ++w) m = fmaxf(m, sm[w]);
  float sum = 0.f;
  for (int i = tid; i < L; i += 256) sum += __expf(x[i] - m);
  sum = wave_sum(sum);
  if (lane == 0) ss[wave] = sum;
  __syncthreads();
  float tot = 0.f;
#pragma unroll
  for (int w = 0; w < 8; ++w) tot += ss[w];
  const float inv = 1.0f / tot;
  for (int i = tid; i < L; i += 256) y[i] = (bf16)(__expf(x[i] - m) * inv);
}

// ---------------- WMMA GEMM core: 128x128 block tile -----------------
// A  : [M,K] row-major bf16;  Bt : [N,K] K-major bf16 (B transposed).
// 8 waves in 4(M) x 2(N); each wave: 32(M) x 64(N) = 2x4 accumulators.
// Fragment layouts per CDNA5 ISA 7.12.2:
//   A 16x32: lane<16 holds K 0..7/16..23, lane>=16 holds K 8..15/24..31
//   B 32x16: lane&15 = column, lanes<16 K 0..7/16..23, lanes>=16 K 8..15/24..31
__device__ __forceinline__ void gemm_core_128x128(
    const bf16* __restrict__ A, long lda, const bf16* __restrict__ Bt,
    long ldb, int m0, int n0, int K, bf16* As, bf16* Bs, v8f (&acc)[2][4]) {
  const int tid  = threadIdx.x;
  const int lane = tid & 31;
  const int wave = tid >> 5;
  const int wm   = wave >> 1;   // 0..3 -> m offset wm*32
  const int wn   = wave & 1;    // 0..1 -> n offset wn*64
  const int r    = lane & 15;
  const int kg   = lane >> 4;

  // staging: 128 rows x 32 K = 512 b128 chunks; 2 chunks/thread
  const int ar0 = tid >> 1;                 // 0..127
  const int ao0 = (tid & 1) * 16;           // 0 or 16  (chunk pair per row)
  const int ar1 = ar0;                      // second chunk: same row, next 8 K (same half)
  const int ao1 = ao0 ^ 8;                  // interleave 8-elem segments

  // Per-thread: row ar0, segments ao0 and ao1 (each 8 bf16).
  v8bf pa0, pa1, pb0, pb1;
  pa0 = *(const v8bf*)&A[(size_t)(m0 + ar0) * lda + ao0];
  pa1 = *(const v8bf*)&A[(size_t)(m0 + ar1) * lda + ao1];
  pb0 = *(const v8bf*)&Bt[(size_t)(n0 + ar0) * ldb + ao0];
  pb1 = *(const v8bf*)&Bt[(size_t)(n0 + ar1) * ldb + ao1];

  int buf = 0;
  for (int kb = 0; kb < K; kb += 32) {
    bf16* as = As + buf * TILE_ELEMS;
    bf16* bs = Bs + buf * TILE_ELEMS;
    *(v8bf*)&as[ar0 * PAD_LD + ao0] = pa0;
    *(v8bf*)&as[ar1 * PAD_LD + ao1] = pa1;
    *(v8bf*)&bs[ar0 * PAD_LD + ao0] = pb0;
    *(v8bf*)&bs[ar1 * PAD_LD + ao1] = pb1;
    __syncthreads();
    if (kb + 32 < K) {  // prefetch next K-slab while computing this one
      const int kn = kb + 32;
      pa0 = *(const v8bf*)&A[(size_t)(m0 + ar0) * lda + kn + ao0];
      pa1 = *(const v8bf*)&A[(size_t)(m0 + ar1) * lda + kn + ao1];
      pb0 = *(const v8bf*)&Bt[(size_t)(n0 + ar0) * ldb + kn + ao0];
      pb1 = *(const v8bf*)&Bt[(size_t)(n0 + ar1) * ldb + kn + ao1];
    }
    v16bf af[2], bfr[4];
#pragma unroll
    for (int mi = 0; mi < 2; ++mi) {
      const bf16* p = &as[(wm * 32 + mi * 16 + r) * PAD_LD];
      af[mi] = cat8(*(const v8bf*)&p[kg * 8], *(const v8bf*)&p[16 + kg * 8]);
    }
#pragma unroll
    for (int ni = 0; ni < 4; ++ni) {
      const bf16* p = &bs[(wn * 64 + ni * 16 + r) * PAD_LD];
      bfr[ni] = cat8(*(const v8bf*)&p[kg * 8], *(const v8bf*)&p[16 + kg * 8]);
    }
#pragma unroll
    for (int mi = 0; mi < 2; ++mi)
#pragma unroll
      for (int ni = 0; ni < 4; ++ni)
        acc[mi][ni] = __builtin_amdgcn_wmma_f32_16x16x32_bf16(
            false, af[mi], false, bfr[ni], (short)0, acc[mi][ni], false, false);
    buf ^= 1;
  }
}

// ---------------- Generic batched GEMM: C = A x Bt^T -----------------
__global__ __launch_bounds__(256) void gemm_bf16_kernel(
    const bf16* __restrict__ A, long lda, long bsA,
    const bf16* __restrict__ Bt, long ldb, long bsB,
    float* __restrict__ C, long ldc, long bsC, int K) {
  __shared__ __align__(128) bf16 As[2 * TILE_ELEMS];
  __shared__ __align__(128) bf16 Bs[2 * TILE_ELEMS];
  const int b  = blockIdx.z;
  const int m0 = blockIdx.y * 128;
  const int n0 = blockIdx.x * 128;
  v8f acc[2][4] = {};
  gemm_core_128x128(A + (size_t)b * bsA, lda, Bt + (size_t)b * bsB, ldb, m0,
                    n0, K, As, Bs, acc);
  const int lane  = threadIdx.x & 31;
  const int wave  = threadIdx.x >> 5;
  const int wm    = wave >> 1, wn = wave & 1;
  const int rbase = (lane >> 4) * 8;
  const int ncol  = lane & 15;
  float* Cp = C + (size_t)b * bsC;
#pragma unroll
  for (int mi = 0; mi < 2; ++mi) {
    const int mrow0 = m0 + wm * 32 + mi * 16 + rbase;
#pragma unroll
    for (int ni = 0; ni < 4; ++ni) {
      const int n = n0 + wn * 64 + ni * 16 + ncol;
#pragma unroll
      for (int rr = 0; rr < 8; ++rr)
        Cp[(size_t)(mrow0 + rr) * ldc + n] = acc[mi][ni][rr];
    }
  }
}

// ---------------- QKV GEMM with fused strided q/k/v extraction --------------
// qkv[m,n], n in [0,1536): channel e = n/3, sel = n%3 (torch reshape(B,N,E,3)).
// q -> Q[m,e] (f32, rows for channel softmax)
// k -> KT[b,e,tok] (f32, rows for token softmax)
// v -> VT[b,e,tok] (bf16, K-major operand for kv GEMM)
__global__ __launch_bounds__(256) void gemm_qkv_kernel(
    const bf16* __restrict__ H,      // [32768, 512]
    const bf16* __restrict__ WqkvT,  // [1536, 512] K-major
    float* __restrict__ Q, float* __restrict__ KT, bf16* __restrict__ VT) {
  __shared__ __align__(128) bf16 As[2 * TILE_ELEMS];
  __shared__ __align__(128) bf16 Bs[2 * TILE_ELEMS];
  const int m0 = blockIdx.y * 128;
  const int n0 = blockIdx.x * 128;
  v8f acc[2][4] = {};
  gemm_core_128x128(H, 512, WqkvT, 512, m0, n0, 512, As, Bs, acc);
  const int lane  = threadIdx.x & 31;
  const int wave  = threadIdx.x >> 5;
  const int wm    = wave >> 1, wn = wave & 1;
  const int rbase = (lane >> 4) * 8;
  const int ncol  = lane & 15;
#pragma unroll
  for (int mi = 0; mi < 2; ++mi) {
    const int mrow0 = m0 + wm * 32 + mi * 16 + rbase;
#pragma unroll
    for (int ni = 0; ni < 4; ++ni) {
      const int n   = n0 + wn * 64 + ni * 16 + ncol;
      const int e   = n / 3;
      const int sel = n - e * 3;
#pragma unroll
      for (int rr = 0; rr < 8; ++rr) {
        const float val = acc[mi][ni][rr];
        const int m   = mrow0 + rr;
        const int bb  = m >> 12;  // / 4096 tokens
        const int tok = m & 4095;
        if (sel == 0) {
          Q[(size_t)m * 512 + e] = val;
        } else {
          const size_t idx = ((size_t)bb * 512 + e) * 4096 + tok;
          if (sel == 1) KT[idx] = val;
          else          VT[idx] = (bf16)val;
        }
      }
    }
  }
}

// ---------------- Weight / kv transposition (f32 -> bf16 K-major) -----------
__global__ void convert_wqkv_kernel(const float* __restrict__ W,
                                    bf16* __restrict__ Wt) {
  const int i = blockIdx.x * 256 + threadIdx.x;  // over 1536*512
  if (i >= 1536 * 512) return;
  const int nn = i >> 9, kk = i & 511;
  Wt[i] = (bf16)W[(size_t)kk * 1536 + nn];
}
__global__ void convert_wproj_kernel(const float* __restrict__ W,
                                     bf16* __restrict__ Wt) {
  const int i = blockIdx.x * 256 + threadIdx.x;  // over 512*512
  if (i >= 512 * 512) return;
  const int dd = i >> 9, ee = i & 511;
  Wt[i] = (bf16)W[(size_t)ee * 512 + dd];
}
__global__ void transpose_kv_kernel(const float* __restrict__ KV,
                                    bf16* __restrict__ KVT) {
  const int i = blockIdx.x * 256 + threadIdx.x;  // over 8*512*512
  if (i >= 8 * 512 * 512) return;
  const int bb  = i >> 18;
  const int rem = i & ((1 << 18) - 1);
  const int ff  = rem >> 9;
  const int ee  = rem & 511;
  KVT[i] = (bf16)KV[((size_t)bb << 18) + ((size_t)ee << 9) + ff];
}

// ---------------------------------------------------------------------------
extern "C" void kernel_launch(void* const* d_in, const int* in_sizes, int n_in,
                              void* d_out, int out_size, void* d_ws,
                              size_t ws_size, hipStream_t stream) {
  (void)in_sizes; (void)n_in; (void)out_size; (void)ws_size;
  const float* x        = (const float*)d_in[0];
  const float* w_qkv    = (const float*)d_in[1];
  const float* w_proj   = (const float*)d_in[2];
  const float* ln_in_g  = (const float*)d_in[3];
  const float* ln_in_b  = (const float*)d_in[4];
  const float* ln_out_g = (const float*)d_in[5];
  const float* ln_out_b = (const float*)d_in[6];
  float* out = (float*)d_out;

  const int Bn = 8, Ntok = 4096, D = 512, E = 512, TE = 1536;
  const int M = Bn * Ntok;  // 32768

  char* ws = (char*)d_ws;
  size_t off = 0;
  auto take = [&](size_t bytes) -> char* {
    char* p = ws + off;
    off += (bytes + 255) & ~(size_t)255;
    return p;
  };
  bf16*  wqkvT  = (bf16*)take((size_t)TE * D * 2);
  bf16*  wprojT = (bf16*)take((size_t)D * E * 2);
  bf16*  hb     = (bf16*)take((size_t)M * D * 2);          // reused as a_ln
  float* qf     = (float*)take((size_t)M * E * 4);         // reused as attn
  float* kT     = (float*)take((size_t)Bn * E * Ntok * 4);
  bf16*  vT     = (bf16*)take((size_t)Bn * E * Ntok * 2);
  bf16*  phiq   = (bf16*)take((size_t)M * E * 2);
  bf16*  phikT  = (bf16*)take((size_t)Bn * E * Ntok * 2);
  float* kv     = (float*)take((size_t)Bn * E * E * 4);
  bf16*  kvT    = (bf16*)take((size_t)Bn * E * E * 2);

  // 0) weight layout conversion (f32 -> bf16, K-major)
  convert_wqkv_kernel<<<(TE * D + 255) / 256, 256, 0, stream>>>(w_qkv, wqkvT);
  convert_wproj_kernel<<<(D * E + 255) / 256, 256, 0, stream>>>(w_proj, wprojT);

  // 1) input LayerNorm -> bf16
  ln_bf16_kernel<<<M, 256, 0, stream>>>(x, ln_in_g, ln_in_b, hb, D);

  // 2) qkv GEMM + fused extraction/transposition
  gemm_qkv_kernel<<<dim3(TE / 128, M / 128), 256, 0, stream>>>(hb, wqkvT, qf,
                                                               kT, vT);

  // 3) softmaxes: phi_q over channels (rows of qf), phi_k over tokens (kT)
  softmax_bf16_kernel<<<M, 256, 0, stream>>>(qf, phiq, E);
  softmax_bf16_kernel<<<Bn * E, 256, 0, stream>>>(kT, phikT, Ntok);

  // 4) kv[b] = phi_kT [E,N] x v [N,E]   (Bt = vT, K-major over tokens)
  gemm_bf16_kernel<<<dim3(E / 128, E / 128, Bn), 256, 0, stream>>>(
      phikT, Ntok, (long)E * Ntok, vT, Ntok, (long)E * Ntok,
      kv, E, (long)E * E, Ntok);
  transpose_kv_kernel<<<(Bn * E * E + 255) / 256, 256, 0, stream>>>(kv, kvT);

  // 5) attn[b] = phi_q [N,E] x kv [E,E]  (Bt = kvT), reuse qf as attn
  float* attn = qf;
  gemm_bf16_kernel<<<dim3(E / 128, Ntok / 128, Bn), 256, 0, stream>>>(
      phiq, E, (long)Ntok * E, kvT, E, (long)E * E,
      attn, E, (long)Ntok * E, E);

  // 6) output LayerNorm -> bf16 (reuse hb), final projection into d_out
  bf16* aln = hb;
  ln_bf16_kernel<<<M, 256, 0, stream>>>(attn, ln_out_g, ln_out_b, aln, E);
  gemm_bf16_kernel<<<dim3(D / 128, M / 128, 1), 256, 0, stream>>>(
      aln, E, 0, wprojT, E, 0, out, D, 0, E);
}
